// MLADecoderCrossAttention_73787538145766
// MI455X (gfx1250) — matrix operand order repaired and checked by reading, not compiled, not run on hardware
//
#include <hip/hip_runtime.h>

// ---------------------------------------------------------------------------
// MLA decoder cross-attention for gfx1250 (MI455X).
// All matmuls via v_wmma_f32_16x16x32_bf16 (wave32); GEMM tiles staged into
// LDS by the Tensor Data Mover (tensor_load_to_lds, TENSORcnt) when the
// builtin is available, with double buffering to overlap DMA and WMMA.
// ---------------------------------------------------------------------------

typedef __attribute__((ext_vector_type(16))) __bf16 v16bf;
typedef __attribute__((ext_vector_type(8)))  __bf16 v8bf;
typedef __attribute__((ext_vector_type(8)))  float  v8f;

#define B_   4
#define SQ_  2048
#define SK_  4096
#define H_   16
#define DH_  64
#define DD_  1024
#define R_   256

static __device__ __forceinline__ v16bf cat8(v8bf a, v8bf b) {
  return __builtin_shufflevector(a, b, 0,1,2,3,4,5,6,7,8,9,10,11,12,13,14,15);
}

static __device__ __forceinline__ v8f wmma_bf16(v16bf a, v16bf b, v8f c) {
  // D = A(16x32) * B(32x16) + C, fp32 accumulate
  return __builtin_amdgcn_wmma_f32_16x16x32_bf16(false, a, false, b,
                                                 (short)0, c, false, false);
}

// ---------------------------------------------------------------------------
// Tensor Data Mover: 2-D tile load (bf16 elements) global -> LDS.
// D# packing per CDNA5 ISA 8.3/8.4: group0 {count=1, lds_addr, global_addr,
// type=2}; group1 {data_size=2B, tensor dims == tile dims (no OOB), tile
// dims, dim0 row stride}. Groups 2/3 zero (<=2D tensor).
// ---------------------------------------------------------------------------
#if __has_builtin(__builtin_amdgcn_tensor_load_to_lds) && \
    __has_builtin(__builtin_amdgcn_s_wait_tensorcnt)
#define HAVE_TDM 1
#else
#define HAVE_TDM 0
#endif

#if HAVE_TDM
typedef __attribute__((ext_vector_type(4))) unsigned int tdm_v4u;
typedef __attribute__((ext_vector_type(8))) int          tdm_v8i;
typedef __attribute__((ext_vector_type(4))) int          tdm_v4i;

static __device__ __forceinline__ void tdm_load_2d_bf16(unsigned lds_off,
                                                        const __bf16* gptr,
                                                        unsigned tile_w,     // elems per row
                                                        unsigned tile_h,     // rows
                                                        unsigned stride_el)  // row stride, elems
{
  unsigned long long ga = (unsigned long long)(const void*)gptr;
  tdm_v4u g0;
  g0[0] = 1u;                                              // count=1, user desc
  g0[1] = lds_off;                                         // lds_addr [63:32]
  g0[2] = (unsigned)ga;                                    // global_addr lo
  g0[3] = ((unsigned)(ga >> 32) & 0x01ffffffu) | (2u << 30); // addr hi, type=2
  tdm_v8i g1;
  g1[0] = (int)(1u << 16);                                 // wg_mask=0, data_size=2B
  g1[1] = (int)((tile_w & 0xffffu) << 16);                 // tensor_dim0[15:0]
  g1[2] = (int)((tile_w >> 16) | ((tile_h & 0xffffu) << 16)); // d0 hi | d1 lo
  g1[3] = (int)((tile_h >> 16) | (tile_w << 16));          // d1 hi | tile_dim0
  g1[4] = (int)(tile_h & 0xffffu);                         // tile_dim1 (dim2=0)
  g1[5] = (int)stride_el;                                  // dim0_stride[31:0]
  g1[6] = 0;                                               // stride hi / dim1_stride
  g1[7] = 0;
  tdm_v4i z4 = {0, 0, 0, 0};
#if defined(__clang_major__) && (__clang_major__ >= 23)
  tdm_v8i z8 = {0, 0, 0, 0, 0, 0, 0, 0};
  __builtin_amdgcn_tensor_load_to_lds(g0, g1, z4, z4, z8, 0);
#else
  __builtin_amdgcn_tensor_load_to_lds(g0, g1, z4, z4, 0);
#endif
}

static __device__ __forceinline__ unsigned lds_offset_of(const void* p) {
  return (unsigned)(unsigned long long)p;  // low 32 bits of flat LDS addr = offset
}
#endif  // HAVE_TDM

// ---------------------------------------------------------------------------
// fp32 -> bf16 conversion (4 elems/thread)
// ---------------------------------------------------------------------------
__global__ __launch_bounds__(256) void cvt_f32_bf16(const float* __restrict__ x,
                                                    __bf16* __restrict__ y,
                                                    size_t n) {
  size_t i = ((size_t)blockIdx.x * 256 + threadIdx.x) * 4;
  if (i + 3 < n) {
    float4 v = *(const float4*)(x + i);
    y[i + 0] = (__bf16)v.x;
    y[i + 1] = (__bf16)v.y;
    y[i + 2] = (__bf16)v.z;
    y[i + 3] = (__bf16)v.w;
  }
}

// ---------------------------------------------------------------------------
// Generic GEMM: C[M,N] = A[M,K] * W[N,K]^T  (A,W bf16 row-major; C fp32/bf16)
// Block tile 128x64, K-step 32, 256 threads = 8 waves; wave w owns rows
// [w*16, w*16+16) x 64 cols (four 16x16 accumulators).
// TDM path: wave 0 DMAs next K-slice tiles into the alternate LDS buffer,
// s_wait_tensorcnt(2) retires the current slice (TDM ops are in-order).
// ---------------------------------------------------------------------------
template<int OUT_BF16>
__global__ __launch_bounds__(256) void gemm_bf16_wmma(const __bf16* __restrict__ A,
                                                      const __bf16* __restrict__ W,
                                                      float*  __restrict__ Cf,
                                                      __bf16* __restrict__ Cb,
                                                      int M, int N, int K) {
#if HAVE_TDM
  __shared__ __align__(32) __bf16 As[2][128 * 32];
  __shared__ __align__(32) __bf16 Bs[2][64 * 32];
#else
  __shared__ __align__(32) __bf16 As[1][128 * 32];
  __shared__ __align__(32) __bf16 Bs[1][64 * 32];
#endif

  const int tid  = threadIdx.x;
  const int wave = tid >> 5;
  const int lane = tid & 31;
  const int m0 = blockIdx.y * 128;
  const int n0 = blockIdx.x * 64;

  const int lm = lane & 15;      // fragment row/col within 16
  const int hi = lane >> 4;      // half-wave select
  const int kb = hi * 8;         // A-fragment K base (ISA 16-bit A layout)

  v8f acc[4];
  for (int i = 0; i < 4; i++)
    for (int r = 0; r < 8; r++) acc[i][r] = 0.0f;

#if HAVE_TDM
  const bool issuer = (wave == 0);
  const int nsteps = K >> 5;
  if (issuer) {
    tdm_load_2d_bf16(lds_offset_of(&As[0][0]), &A[(size_t)m0 * K], 32, 128, (unsigned)K);
    tdm_load_2d_bf16(lds_offset_of(&Bs[0][0]), &W[(size_t)n0 * K], 32, 64,  (unsigned)K);
  }
  for (int s = 0; s < nsteps; s++) {
    const int buf = s & 1;
    if (issuer) {
      if (s + 1 < nsteps) {
        const int k1 = (s + 1) << 5;
        tdm_load_2d_bf16(lds_offset_of(&As[buf ^ 1][0]), &A[(size_t)m0 * K + k1],
                         32, 128, (unsigned)K);
        tdm_load_2d_bf16(lds_offset_of(&Bs[buf ^ 1][0]), &W[(size_t)n0 * K + k1],
                         32, 64, (unsigned)K);
        __builtin_amdgcn_s_wait_tensorcnt(2);  // current slice (2 oldest) done
      } else {
        __builtin_amdgcn_s_wait_tensorcnt(0);
      }
    }
    __syncthreads();  // current buffer visible to all waves

    v16bf afrag = *(const v16bf*)&As[buf][(wave * 16 + lm) * 32 + hi * 16];
#pragma unroll
    for (int nb = 0; nb < 4; nb++) {
      v16bf bfrag = *(const v16bf*)&Bs[buf][(nb * 16 + lm) * 32 + hi * 16];
      acc[nb] = wmma_bf16(afrag, bfrag, acc[nb]);
    }
    __syncthreads();  // all reads of this buffer done before it is re-DMAed
  }
#else
  const int aRow = tid >> 1, aCol = (tid & 1) * 16;
  const int bRow = tid >> 2, bCol = (tid & 3) * 8;
  for (int k0 = 0; k0 < K; k0 += 32) {
    __syncthreads();
    *(v16bf*)&As[0][aRow * 32 + aCol] =
        *(const v16bf*)&A[(size_t)(m0 + aRow) * K + k0 + aCol];
    *(v8bf*)&Bs[0][bRow * 32 + bCol] =
        *(const v8bf*)&W[(size_t)(n0 + bRow) * K + k0 + bCol];
    __syncthreads();
    if (k0 + 32 < K) {
      __builtin_prefetch(&A[(size_t)(m0 + aRow) * K + k0 + 32 + aCol], 0, 1);
      __builtin_prefetch(&W[(size_t)(n0 + bRow) * K + k0 + 32 + bCol], 0, 1);
    }
    v16bf afrag = *(const v16bf*)&As[0][(wave * 16 + lm) * 32 + hi * 16];
#pragma unroll
    for (int nb = 0; nb < 4; nb++) {
      v16bf bfrag = *(const v16bf*)&Bs[0][(nb * 16 + lm) * 32 + hi * 16];
      acc[nb] = wmma_bf16(afrag, bfrag, acc[nb]);
    }
  }
#endif

  // C/D layout: element r <-> row (r + 8*hi), col = lane&15
#pragma unroll
  for (int nb = 0; nb < 4; nb++)
    for (int r = 0; r < 8; r++) {
      size_t idx = (size_t)(m0 + wave * 16 + r + hi * 8) * N + n0 + nb * 16 + lm;
      if (OUT_BF16) Cb[idx] = (__bf16)acc[nb][r];
      else          Cf[idx] = acc[nb][r];
    }
}

// ---------------------------------------------------------------------------
// LayerNorm over R=256 per row, fp32 in -> bf16 out
// ---------------------------------------------------------------------------
__global__ __launch_bounds__(256) void layernorm_r256(const float* __restrict__ c,
                                                      const float* __restrict__ g,
                                                      const float* __restrict__ b,
                                                      __bf16* __restrict__ out) {
  const int row = blockIdx.x, t = threadIdx.x;
  __shared__ float red[256];
  float v = c[(size_t)row * R_ + t];
  red[t] = v;
  __syncthreads();
  for (int s = 128; s > 0; s >>= 1) {
    if (t < s) red[t] += red[t + s];
    __syncthreads();
  }
  float mu = red[0] * (1.0f / R_);
  __syncthreads();
  float d = v - mu;
  red[t] = d * d;
  __syncthreads();
  for (int s = 128; s > 0; s >>= 1) {
    if (t < s) red[t] += red[t + s];
    __syncthreads();
  }
  float var = red[0] * (1.0f / R_);
  float rs = rsqrtf(var + 1e-5f);
  out[(size_t)row * R_ + t] = (__bf16)(d * rs * g[t] + b[t]);
}

// ---------------------------------------------------------------------------
// Flash attention: block = 4 waves (128 thr), each wave owns 16 query rows,
// block covers 64 query rows of one (b,h). Streams SK in 64-key tiles
// (16 WMMA per tile; softmax butterfly cost amortized over 2x more keys).
// q: (B,SQ,H*DH) bf16 ; kv: (B,SK,H*128) bf16 (k first 64, v last 64)
// o: (B,SQ,H*DH) bf16
// ---------------------------------------------------------------------------
__global__ __launch_bounds__(128) void flash_attn_wmma(const __bf16* __restrict__ q,
                                                       const __bf16* __restrict__ kv,
                                                       __bf16* __restrict__ o) {
  const int bh = blockIdx.y;
  const int b = bh >> 4, h = bh & 15;
  const int q0 = blockIdx.x * 64;
  const int tid = threadIdx.x;
  const int wave = tid >> 5, lane = tid & 31;
  const int lm = lane & 15, hi = lane >> 4, kb = hi * 8;

  __shared__ __align__(32) __bf16 Ks[64][64];      // 64 keys x 64 dims
  __shared__ __align__(32) __bf16 Vt[64][64];      // transposed: dim x key
  __shared__ __align__(32) __bf16 Ps[4][16][64];   // per-wave P staging

  // Q fragments (fixed over the SK loop): 16x64 = two 16x32 A-fragments
  const __bf16* qrow =
      q + ((size_t)(b * SQ_ + q0 + wave * 16 + lm)) * (H_ * DH_) + h * DH_;
  v16bf qa[2];
#pragma unroll
  for (int kc = 0; kc < 2; kc++) {
    qa[kc] = *(const v16bf*)(qrow + kc * 32 + hi * 16);
  }

  float rmax[8], rsum[8];
  for (int r = 0; r < 8; r++) { rmax[r] = -1e30f; rsum[r] = 0.0f; }
  v8f oacc[4];
  for (int i = 0; i < 4; i++)
    for (int r = 0; r < 8; r++) oacc[i][r] = 0.0f;

  const float scale = 0.125f;  // 1/sqrt(64)
  const size_t kvbase = ((size_t)b * SK_) * (H_ * 2 * DH_) + h * (2 * DH_);

  for (int sk0 = 0; sk0 < SK_; sk0 += 64) {
    __syncthreads();  // previous PV reads of Ks/Vt complete
    {  // cooperative K/V tile load; V stored transposed
      int key = tid >> 1;
      int dch = (tid & 1) * 32;
      const __bf16* src = kv + kvbase + (size_t)(sk0 + key) * (H_ * 2 * DH_) + dch;
      *(v16bf*)&Ks[key][dch]      = *(const v16bf*)src;
      *(v16bf*)&Ks[key][dch + 16] = *(const v16bf*)(src + 16);
      v16bf v0 = *(const v16bf*)(src + DH_);
      v16bf v1 = *(const v16bf*)(src + DH_ + 16);
#pragma unroll
      for (int i = 0; i < 16; i++) {
        Vt[dch + i][key]      = v0[i];
        Vt[dch + 16 + i][key] = v1[i];
      }
      if (sk0 + 64 < SK_)
        __builtin_prefetch(src + (size_t)64 * (H_ * 2 * DH_), 0, 1);
    }
    __syncthreads();

    // S(16x64) = Q(16x64) @ K^T : four n-blocks, each sums two K=32 chunks
    v8f s[4];
#pragma unroll
    for (int nb = 0; nb < 4; nb++) {
      v8f a;
      for (int r = 0; r < 8; r++) a[r] = 0.0f;
#pragma unroll
      for (int kc = 0; kc < 2; kc++) {
        v16bf bfrag = *(const v16bf*)&Ks[nb * 16 + lm][kc * 32 + hi * 16];
        a = wmma_bf16(qa[kc], bfrag, a);
      }
      s[nb] = a;
    }

    // online softmax (rows striped: element r <-> row r + 8*hi; cols = lanes%16)
    float mnew[8];
#pragma unroll
    for (int r = 0; r < 8; r++) {
      float m01 = fmaxf(s[0][r] * scale, s[1][r] * scale);
      float m23 = fmaxf(s[2][r] * scale, s[3][r] * scale);
      s[0][r] *= scale; s[1][r] *= scale; s[2][r] *= scale; s[3][r] *= scale;
      mnew[r] = fmaxf(rmax[r], fmaxf(m01, m23));
    }
#pragma unroll
    for (int r = 0; r < 8; r++)
      for (int off = 1; off < 16; off <<= 1)
        mnew[r] = fmaxf(mnew[r], __shfl_xor(mnew[r], off, 32));

    float alpha[8], psum[8];
#pragma unroll
    for (int r = 0; r < 8; r++) {
      alpha[r] = __expf(rmax[r] - mnew[r]);
      float p0 = __expf(s[0][r] - mnew[r]);
      float p1 = __expf(s[1][r] - mnew[r]);
      float p2 = __expf(s[2][r] - mnew[r]);
      float p3 = __expf(s[3][r] - mnew[r]);
      psum[r] = (p0 + p1) + (p2 + p3);
      int row = r + hi * 8;
      Ps[wave][row][lm]      = (__bf16)p0;
      Ps[wave][row][16 + lm] = (__bf16)p1;
      Ps[wave][row][32 + lm] = (__bf16)p2;
      Ps[wave][row][48 + lm] = (__bf16)p3;
      rmax[r] = mnew[r];
    }
#pragma unroll
    for (int r = 0; r < 8; r++)
      for (int off = 1; off < 16; off <<= 1)
        psum[r] += __shfl_xor(psum[r], off, 32);
#pragma unroll
    for (int r = 0; r < 8; r++) rsum[r] = rsum[r] * alpha[r] + psum[r];
#pragma unroll
    for (int nb = 0; nb < 4; nb++)
      for (int r = 0; r < 8; r++) oacc[nb][r] *= alpha[r];

    __syncthreads();  // Ps visible; fences before this wave's PV fragment loads

    // P A-fragments (16x64 = two 16x32), then O += P @ V
    v16bf pfrag[2];
#pragma unroll
    for (int kc = 0; kc < 2; kc++) {
      pfrag[kc] = *(const v16bf*)&Ps[wave][lm][kc * 32 + hi * 16];
    }
#pragma unroll
    for (int nb = 0; nb < 4; nb++) {
#pragma unroll
      for (int kc = 0; kc < 2; kc++) {
        v16bf vfrag = *(const v16bf*)&Vt[nb * 16 + lm][kc * 32 + hi * 16];
        oacc[nb] = wmma_bf16(pfrag[kc], vfrag, oacc[nb]);
      }
    }
  }

  // normalize and store o (bf16, feeds out-proj GEMM)
  float inv[8];
  for (int r = 0; r < 8; r++) inv[r] = 1.0f / rsum[r];
#pragma unroll
  for (int nb = 0; nb < 4; nb++)
    for (int r = 0; r < 8; r++) {
      size_t idx = (size_t)(b * SQ_ + q0 + wave * 16 + r + hi * 8) * (H_ * DH_) +
                   h * DH_ + nb * 16 + lm;
      o[idx] = (__bf16)(oacc[nb][r] * inv[r]);
    }
}

// ---------------------------------------------------------------------------
// Host-side orchestration
// ---------------------------------------------------------------------------
extern "C" void kernel_launch(void* const* d_in, const int* in_sizes, int n_in,
                              void* d_out, int out_size, void* d_ws, size_t ws_size,
                              hipStream_t stream) {
  const float* query = (const float*)d_in[0];   // (4,2048,1024)
  const float* enc   = (const float*)d_in[1];   // (4,4096,1024)
  const float* Wq    = (const float*)d_in[2];   // (1024,1024)
  const float* Wkva  = (const float*)d_in[3];   // (256,1024)
  const float* ln_g  = (const float*)d_in[4];   // (256,)
  const float* ln_b  = (const float*)d_in[5];   // (256,)
  const float* Wkvb  = (const float*)d_in[6];   // (2048,256)
  const float* Wout  = (const float*)d_in[7];   // (1024,1024)
  float* out = (float*)d_out;                   // (4,2048,1024)

  const size_t N_query = (size_t)B_ * SQ_ * DD_;       // 8M
  const size_t N_enc   = (size_t)B_ * SK_ * DD_;       // 16M
  const size_t N_Wq    = (size_t)DD_ * DD_;
  const size_t N_Wkva  = (size_t)R_ * DD_;
  const size_t N_Wkvb  = (size_t)H_ * 2 * DH_ * R_;
  const size_t N_Wout  = (size_t)DD_ * DD_;
  const size_t N_c     = (size_t)B_ * SK_ * R_;
  const size_t N_kv    = (size_t)B_ * SK_ * H_ * 2 * DH_;

  __bf16* p = (__bf16*)d_ws;
  __bf16* query_bf = p; p += N_query;
  __bf16* enc_bf   = p; p += N_enc;
  __bf16* Wq_bf    = p; p += N_Wq;
  __bf16* Wkva_bf  = p; p += N_Wkva;
  __bf16* Wkvb_bf  = p; p += N_Wkvb;
  __bf16* Wout_bf  = p; p += N_Wout;
  __bf16* q_bf     = p; p += N_query;
  __bf16* c_bf     = p; p += N_c;
  __bf16* kv_bf    = p; p += N_kv;
  __bf16* o_bf     = p; p += N_query;
  float*  c_f      = (float*)p;  // 16384*256 fp32

  // 1) fp32 -> bf16
  cvt_f32_bf16<<<dim3(N_query / 1024), 256, 0, stream>>>(query, query_bf, N_query);
  cvt_f32_bf16<<<dim3(N_enc   / 1024), 256, 0, stream>>>(enc,   enc_bf,   N_enc);
  cvt_f32_bf16<<<dim3(N_Wq    / 1024), 256, 0, stream>>>(Wq,    Wq_bf,    N_Wq);
  cvt_f32_bf16<<<dim3(N_Wkva  / 1024), 256, 0, stream>>>(Wkva,  Wkva_bf,  N_Wkva);
  cvt_f32_bf16<<<dim3(N_Wkvb  / 1024), 256, 0, stream>>>(Wkvb,  Wkvb_bf,  N_Wkvb);
  cvt_f32_bf16<<<dim3(N_Wout  / 1024), 256, 0, stream>>>(Wout,  Wout_bf,  N_Wout);

  // 2) q = query @ Wq^T   (8192x1024x1024) -> bf16
  gemm_bf16_wmma<1><<<dim3(1024 / 64, 8192 / 128), 256, 0, stream>>>(
      query_bf, Wq_bf, nullptr, q_bf, 8192, 1024, 1024);

  // 3) c = enc @ Wkva^T   (16384x256x1024) -> fp32 (for exact-ish LN stats)
  gemm_bf16_wmma<0><<<dim3(256 / 64, 16384 / 128), 256, 0, stream>>>(
      enc_bf, Wkva_bf, c_f, nullptr, 16384, 256, 1024);

  // 4) layernorm(R=256) -> bf16
  layernorm_r256<<<dim3(16384), 256, 0, stream>>>(c_f, ln_g, ln_b, c_bf);

  // 5) kv = c @ Wkvb^T    (16384x2048x256) -> bf16
  gemm_bf16_wmma<1><<<dim3(2048 / 64, 16384 / 128), 256, 0, stream>>>(
      c_bf, Wkvb_bf, nullptr, kv_bf, 16384, 2048, 256);

  // 6) flash attention -> o_bf  (grid: SQ/64 x B*H)
  flash_attn_wmma<<<dim3(SQ_ / 64, B_ * H_), 128, 0, stream>>>(q_bf, kv_bf, o_bf);

  // 7) out = o @ Wout^T   (8192x1024x1024) -> fp32 d_out
  gemm_bf16_wmma<0><<<dim3(1024 / 64, 8192 / 128), 256, 0, stream>>>(
      o_bf, Wout_bf, out, nullptr, 8192, 1024, 1024);

  (void)in_sizes; (void)n_in; (void)out_size; (void)ws_size;
}
